// PQAdapter_6038724018865
// MI455X (gfx1250) — compile-verified, matrix operand order checked by reading, not run on hardware
//
#include <hip/hip_runtime.h>
#include <math.h>

// ---------------------------------------------------------------------------
// CDNA5 (gfx1250) implementation of the PQAdapter forward pass.
// Heavy GEMMs (cosine-cost matmul, conv1/conv2 implicit GEMM) run on
// v_wmma_f32_16x16x32_bf16; everything else is lightweight VALU work.
// ---------------------------------------------------------------------------

typedef __attribute__((ext_vector_type(16))) __bf16 v16bf;
typedef __attribute__((ext_vector_type(8)))  float  v8f;
typedef __attribute__((ext_vector_type(4)))  unsigned int v4u;

union Frag { v16bf v; v4u q[2]; };

__device__ __forceinline__ unsigned short f2bf(float f) {
    unsigned int u = __float_as_uint(f);
    unsigned int r = u + 0x7FFFu + ((u >> 16) & 1u);
    return (unsigned short)(r >> 16);
}
__device__ __forceinline__ float bf2f(unsigned short h) {
    return __uint_as_float(((unsigned int)h) << 16);
}

// ---------------------------------------------------------------------------
// Weight transform: (O,C,3,3) f32 -> (O, 9*C) bf16, k-order = (tap, c)
// ---------------------------------------------------------------------------
__global__ void wtrans_kernel(const float* __restrict__ w,
                              unsigned short* __restrict__ out, int O, int C) {
    int i = blockIdx.x * 256 + threadIdx.x;
    int total = O * C * 9;
    if (i >= total) return;
    int o = i / (C * 9);
    int r = i % (C * 9);
    int tap = r / C;
    int c = r % C;
    int dy = tap / 3, dx = tap % 3;
    out[i] = f2bf(w[(((size_t)o * C + c) * 3 + dy) * 3 + dx]);
}

// ---------------------------------------------------------------------------
// L2-normalize query rows (strip CLS): -> Qbf (B*256, 768) bf16 row-major
// ---------------------------------------------------------------------------
__global__ void qnorm_kernel(const float* __restrict__ qp,
                             unsigned short* __restrict__ Qbf) {
    __shared__ float red[256];
    int blk = blockIdx.x;             // b*256 + m
    int b = blk >> 8, m = blk & 255;
    const float* src = qp + ((size_t)b * 257 + m + 1) * 768;
    int t = threadIdx.x;
    float s = 0.f;
    for (int d = t; d < 768; d += 256) { float v = src[d]; s += v * v; }
    red[t] = s; __syncthreads();
    for (int st = 128; st > 0; st >>= 1) {
        if (t < st) red[t] += red[t + st];
        __syncthreads();
    }
    float inv = 1.f / fmaxf(sqrtf(red[0]), 1e-12f);
    for (int d = t; d < 768; d += 256)
        Qbf[(size_t)blk * 768 + d] = f2bf(src[d] * inv);
}

// ---------------------------------------------------------------------------
// L2-normalize prompt rows (strip CLS, flatten K*256): -> Pbf (B*2048,768) bf16
// ---------------------------------------------------------------------------
__global__ void pnorm_kernel(const float* __restrict__ pp,
                             unsigned short* __restrict__ Pbf) {
    __shared__ float red[256];
    int blk = blockIdx.x;             // b*2048 + j
    int b = blk >> 11, j = blk & 2047;
    int k = j >> 8, pat = j & 255;
    const float* src = pp + (((size_t)b * 8 + k) * 257 + pat + 1) * 768;
    int t = threadIdx.x;
    float s = 0.f;
    for (int d = t; d < 768; d += 256) { float v = src[d]; s += v * v; }
    red[t] = s; __syncthreads();
    for (int st = 128; st > 0; st >>= 1) {
        if (t < st) red[t] += red[t + st];
        __syncthreads();
    }
    float inv = 1.f / fmaxf(sqrtf(red[0]), 1e-12f);
    for (int d = t; d < 768; d += 256)
        Pbf[(size_t)blk * 768 + d] = f2bf(src[d] * inv);
}

// ---------------------------------------------------------------------------
// Cost matmul + fused min/argmin.
// grid = (16 M-tiles, 16 batches), 256 threads (8 waves).
// Each wave sweeps 16 of the 128 p-row tiles in groups of 4: one A-fragment
// LDS load feeds 4 v_wmma_f32_16x16x32_bf16 (4 live accumulators), with an
// opaque-offset barrier so the compiler cannot hoist (and spill) A fragments.
// ---------------------------------------------------------------------------
__global__ void __launch_bounds__(256)
cost_wmma_kernel(const unsigned short* __restrict__ Qbf,
                 const unsigned short* __restrict__ Pbf,
                 float* __restrict__ alignOut,
                 int* __restrict__ idxOut) {
    __shared__ __align__(16) unsigned short Ash[16 * 768];   // 24 KB
    __shared__ float rv[256 * 8];
    __shared__ int   ri[256 * 8];
    int t = threadIdx.x;
    int Mtile = blockIdx.x;
    int b = blockIdx.y;

    // Stage the 16 query rows in LDS (reused by all 128 k-tiles).
    const unsigned int* qsrc =
        (const unsigned int*)(Qbf + ((size_t)(b * 256 + Mtile * 16)) * 768);
    unsigned int* adst = (unsigned int*)Ash;
    for (int i = t; i < 6144; i += 256) adst[i] = qsrc[i];
    __syncthreads();

    int wave = t >> 5;
    int lane = t & 31;
    int arow = lane & 15;
    int sel = (lane >> 4) * 8;        // bf16 A/B fragment K-slice select

    float rmin[8];
    int   ridx[8];
#pragma unroll
    for (int j = 0; j < 8; ++j) { rmin[j] = 3.4e38f; ridx[j] = 0x7fffffff; }

    for (int g = 0; g < 4; ++g) {
        int kt0 = wave * 16 + g * 4;                 // first of 4 k-tiles
        int jrow0 = kt0 * 16 + (lane & 15);          // p-row for tile 0
        const unsigned short* prow =
            Pbf + ((size_t)(b * 2048 + jrow0)) * 768;

        // Opaque offset: stops LICM from hoisting all A fragments out of the
        // g-loop (which previously forced a 192-VGPR spill to scratch).
        int aoff = arow * 768;
        asm volatile("" : "+v"(aoff));
        const unsigned short* asrc = Ash + aoff;

        v8f acc0 = {0.f, 0.f, 0.f, 0.f, 0.f, 0.f, 0.f, 0.f};
        v8f acc1 = acc0, acc2 = acc0, acc3 = acc0;
#pragma unroll 2
        for (int c0 = 0; c0 < 768; c0 += 32) {
            Frag a;
            a.q[0] = *(const v4u*)(asrc + c0 + sel);
            a.q[1] = *(const v4u*)(asrc + c0 + 16 + sel);
            Frag b0, b1, b2, b3;
            b0.q[0] = *(const v4u*)(prow +  0 * 12288 + c0 + sel);
            b0.q[1] = *(const v4u*)(prow +  0 * 12288 + c0 + 16 + sel);
            b1.q[0] = *(const v4u*)(prow +  1 * 12288 + c0 + sel);
            b1.q[1] = *(const v4u*)(prow +  1 * 12288 + c0 + 16 + sel);
            b2.q[0] = *(const v4u*)(prow +  2 * 12288 + c0 + sel);
            b2.q[1] = *(const v4u*)(prow +  2 * 12288 + c0 + 16 + sel);
            b3.q[0] = *(const v4u*)(prow +  3 * 12288 + c0 + sel);
            b3.q[1] = *(const v4u*)(prow +  3 * 12288 + c0 + 16 + sel);
            acc0 = __builtin_amdgcn_wmma_f32_16x16x32_bf16(
                false, a.v, false, b0.v, (short)0, acc0, false, false);
            acc1 = __builtin_amdgcn_wmma_f32_16x16x32_bf16(
                false, a.v, false, b1.v, (short)0, acc1, false, false);
            acc2 = __builtin_amdgcn_wmma_f32_16x16x32_bf16(
                false, a.v, false, b2.v, (short)0, acc2, false, false);
            acc3 = __builtin_amdgcn_wmma_f32_16x16x32_bf16(
                false, a.v, false, b3.v, (short)0, acc3, false, false);
        }
        // min/argmin update, ascending k order keeps first-occurrence ties.
#pragma unroll
        for (int j = 0; j < 8; ++j) {
            float c0v = 1.f - acc0[j];
            if (c0v < rmin[j]) { rmin[j] = c0v; ridx[j] = jrow0; }
            float c1v = 1.f - acc1[j];
            if (c1v < rmin[j]) { rmin[j] = c1v; ridx[j] = jrow0 + 16; }
            float c2v = 1.f - acc2[j];
            if (c2v < rmin[j]) { rmin[j] = c2v; ridx[j] = jrow0 + 32; }
            float c3v = 1.f - acc3[j];
            if (c3v < rmin[j]) { rmin[j] = c3v; ridx[j] = jrow0 + 48; }
        }
    }
#pragma unroll
    for (int j = 0; j < 8; ++j) { rv[t * 8 + j] = rmin[j]; ri[t * 8 + j] = ridx[j]; }
    __syncthreads();

    if (t < 16) {
        int m = t;
        int half = (m >= 8) ? 16 : 0;
        int j = m & 7;
        float bv = 3.4e38f; int bi = 0x7fffffff;
        for (int w = 0; w < 8; ++w)
            for (int ln = 0; ln < 16; ++ln) {
                int t2 = w * 32 + ln + half;
                float v = rv[t2 * 8 + j];
                int ii = ri[t2 * 8 + j];
                if (v < bv || (v == bv && ii < bi)) { bv = v; bi = ii; }
            }
        alignOut[b * 256 + Mtile * 16 + m] = bv;
        idxOut[b * 256 + Mtile * 16 + m] = bi;
    }
}

// ---------------------------------------------------------------------------
// fusion = q + |q - aligned|  (NHWC f32, gather aligned via idx)
// ---------------------------------------------------------------------------
__global__ void fusion_kernel(const unsigned short* __restrict__ Qbf,
                              const unsigned short* __restrict__ Pbf,
                              const int* __restrict__ idxArr,
                              float* __restrict__ fus) {
    size_t i = (size_t)blockIdx.x * 256 + threadIdx.x;
    if (i >= (size_t)16 * 256 * 768) return;
    int d = (int)(i % 768);
    int row = (int)(i / 768);          // b*256 + pix
    int b = row >> 8;
    int k = idxArr[row];
    float q = bf2f(Qbf[i]);
    float a = bf2f(Pbf[((size_t)(b * 2048 + k)) * 768 + d]);
    fus[i] = q + fabsf(q - a);
}

// ---------------------------------------------------------------------------
// BatchNorm stats over NHWC (channel c): scale/shift form.
// ---------------------------------------------------------------------------
__global__ void bnstats_kernel(const float* __restrict__ x, int C, int N,
                               const float* __restrict__ g,
                               const float* __restrict__ bta,
                               float* __restrict__ scale,
                               float* __restrict__ shift) {
    __shared__ float ssum[256], ssq[256];
    int c = blockIdx.x;
    float s = 0.f, q = 0.f;
    for (int i = threadIdx.x; i < N; i += 256) {
        float v = x[(size_t)i * C + c];
        s += v; q += v * v;
    }
    ssum[threadIdx.x] = s; ssq[threadIdx.x] = q; __syncthreads();
    for (int st = 128; st > 0; st >>= 1) {
        if (threadIdx.x < st) {
            ssum[threadIdx.x] += ssum[threadIdx.x + st];
            ssq[threadIdx.x]  += ssq[threadIdx.x + st];
        }
        __syncthreads();
    }
    if (threadIdx.x == 0) {
        float m = ssum[0] / N;
        float var = ssq[0] / N - m * m;
        float sc = g[c] * rsqrtf(var + 1e-5f);
        scale[c] = sc;
        shift[c] = bta[c] - m * sc;
    }
}

__global__ void bnapply_kernel(const float* __restrict__ x,
                               const float* __restrict__ scale,
                               const float* __restrict__ shift,
                               int C, size_t total,
                               unsigned short* __restrict__ obf,
                               float* __restrict__ of32, int relu) {
    size_t i = (size_t)blockIdx.x * 256 + threadIdx.x;
    if (i >= total) return;
    int c = (int)(i % C);
    float v = x[i] * scale[c] + shift[c];
    if (relu) v = fmaxf(v, 0.f);
    if (obf)  obf[i]  = f2bf(v);
    if (of32) of32[i] = v;
}

// ---------------------------------------------------------------------------
// 3x3 SAME conv as implicit GEMM on WMMA.
// in : NHWC bf16 (16, H, H, C); wt : (O, 9*C) bf16 (tap-major); out NHWC f32.
// grid = (B*H*H/16 M-tiles, O/16 N-tiles), 32 threads (one wave).
// ---------------------------------------------------------------------------
__global__ void __launch_bounds__(32)
conv3x3_wmma_kernel(const unsigned short* __restrict__ in,
                    const unsigned short* __restrict__ wt,
                    float* __restrict__ out,
                    int H, int C, int O) {
    int lane = threadIdx.x;
    int Mtile = blockIdx.x, Ntile = blockIdx.y;
    int r = Mtile * 16 + (lane & 15);          // A-fragment row = pixel
    int HW = H * H;
    int b = r / HW;
    int rem = r % HW;
    int y = rem / H, x = rem % H;
    int sel = (lane >> 4) * 8;
    int orow = Ntile * 16 + (lane & 15);       // B-fragment row = out channel
    const unsigned short* wr = wt + (size_t)orow * (C * 9);

    v8f acc = {0.f, 0.f, 0.f, 0.f, 0.f, 0.f, 0.f, 0.f};
    v4u zero = {0u, 0u, 0u, 0u};
    for (int tap = 0; tap < 9; ++tap) {
        int sy = y + tap / 3 - 1;
        int sx = x + tap % 3 - 1;
        bool inb = (sy >= 0 && sy < H && sx >= 0 && sx < H);
        const unsigned short* arow = in + ((size_t)(b * HW + sy * H + sx)) * C;
        const unsigned short* krow = wr + tap * C;
#pragma unroll 4
        for (int c0 = 0; c0 < C; c0 += 32) {
            Frag a, bb;
            if (inb) {
                a.q[0] = *(const v4u*)(arow + c0 + sel);
                a.q[1] = *(const v4u*)(arow + c0 + 16 + sel);
            } else {
                a.q[0] = zero; a.q[1] = zero;
            }
            bb.q[0] = *(const v4u*)(krow + c0 + sel);
            bb.q[1] = *(const v4u*)(krow + c0 + 16 + sel);
            acc = __builtin_amdgcn_wmma_f32_16x16x32_bf16(
                false, a.v, false, bb.v, (short)0, acc, false, false);
        }
    }
    int moff = (lane >> 4) * 8;
    int n = Ntile * 16 + (lane & 15);
#pragma unroll
    for (int j = 0; j < 8; ++j) {
        int mrow = Mtile * 16 + j + moff;
        out[(size_t)mrow * O + n] = acc[j];
    }
}

// ---------------------------------------------------------------------------
// 2x "deconv" (einsum nchw,coab->nohawb) : NHWC f32 -> NHWC bf16
// ---------------------------------------------------------------------------
__global__ void deconv_kernel(const float* __restrict__ x,
                              const float* __restrict__ w,
                              const float* __restrict__ bias,
                              unsigned short* __restrict__ out,
                              int H, int C, int O) {
    size_t i = (size_t)blockIdx.x * 256 + threadIdx.x;
    size_t total = (size_t)16 * 2 * H * 2 * H * O;
    if (i >= total) return;
    int o = (int)(i % O);
    size_t r = i / O;
    int xa = (int)(r % (2 * H)); r /= (2 * H);
    int ya = (int)(r % (2 * H));
    int b = (int)(r / (2 * H));
    int yy = ya >> 1, a = ya & 1, xx = xa >> 1, bb = xa & 1;
    const float* xin = x + ((size_t)(b * H + yy) * H + xx) * C;
    const float* wr = w + ((size_t)o * 2 + a) * 2 + bb;  // stride over c = O*4
    float s = bias[o];
    for (int c = 0; c < C; ++c) s += xin[c] * wr[(size_t)c * O * 4];
    out[i] = f2bf(s);
}

// ---------------------------------------------------------------------------
// 1x1 conv (64 -> 2) + 2-way softmax. in: NHWC bf16 (16,64,64,64).
// ---------------------------------------------------------------------------
__global__ void conv3sm_kernel(const unsigned short* __restrict__ x,
                               const float* __restrict__ w3,
                               const float* __restrict__ b3,
                               float* __restrict__ sm) {
    int i = blockIdx.x * 256 + threadIdx.x;  // b*4096 + pix
    if (i >= 16 * 4096) return;
    const unsigned short* p = x + (size_t)i * 64;
    float s0 = b3[0], s1 = b3[1];
    for (int c = 0; c < 64; ++c) {
        float v = bf2f(p[c]);
        s0 += v * w3[c];
        s1 += v * w3[64 + c];
    }
    float m = fmaxf(s0, s1);
    float e0 = expf(s0 - m), e1 = expf(s1 - m);
    float inv = 1.f / (e0 + e1);
    sm[(size_t)i * 2]     = e0 * inv;
    sm[(size_t)i * 2 + 1] = e1 * inv;
}

// ---------------------------------------------------------------------------
// Half-pixel bilinear resize S x S -> 224 x 224 with generic source strides.
// dst layout: ((b*C + c) * 224 + oy) * 224 + ox
// ---------------------------------------------------------------------------
__global__ void resize_kernel(const float* __restrict__ src,
                              float* __restrict__ dst,
                              int B, int C, int S,
                              long sB, long sC, long sP) {
    size_t i = (size_t)blockIdx.x * 256 + threadIdx.x;
    size_t total = (size_t)B * C * 224 * 224;
    if (i >= total) return;
    int ox = (int)(i % 224);
    size_t r = i / 224;
    int oy = (int)(r % 224); r /= 224;
    int c = (int)(r % C);
    int b = (int)(r / C);
    float scl = (float)S / 224.f;
    float fy = (oy + 0.5f) * scl - 0.5f;
    float fx = (ox + 0.5f) * scl - 0.5f;
    int y0 = (int)floorf(fy); float ty = fy - (float)y0;
    int x0 = (int)floorf(fx); float tx = fx - (float)x0;
    int y1 = y0 + 1, x1 = x0 + 1;
    y0 = y0 < 0 ? 0 : (y0 > S - 1 ? S - 1 : y0);
    y1 = y1 < 0 ? 0 : (y1 > S - 1 ? S - 1 : y1);
    x0 = x0 < 0 ? 0 : (x0 > S - 1 ? S - 1 : x0);
    x1 = x1 < 0 ? 0 : (x1 > S - 1 ? S - 1 : x1);
    const float* base = src + (size_t)b * sB + (size_t)c * sC;
    float v00 = base[(size_t)(y0 * S + x0) * sP];
    float v01 = base[(size_t)(y0 * S + x1) * sP];
    float v10 = base[(size_t)(y1 * S + x0) * sP];
    float v11 = base[(size_t)(y1 * S + x1) * sP];
    dst[i] = v00 * (1.f - ty) * (1.f - tx) + v01 * (1.f - ty) * tx +
             v10 * ty * (1.f - tx) + v11 * ty * tx;
}

// ---------------------------------------------------------------------------
// pooled[b,d] = (mean_pix + mean(top10_pix)) / 2 over BN'd fusion (bf16 NHWC)
// ---------------------------------------------------------------------------
__global__ void pooled_kernel(const unsigned short* __restrict__ fusbn,
                              float* __restrict__ pooled) {
    int i = blockIdx.x * 256 + threadIdx.x;
    if (i >= 16 * 768) return;
    int b = i / 768, d = i % 768;
    const unsigned short* p = fusbn + (size_t)b * 256 * 768 + d;
    float top[10];
#pragma unroll
    for (int j = 0; j < 10; ++j) top[j] = -3.4e38f;
    float s = 0.f;
    for (int pix = 0; pix < 256; ++pix) {
        float v = bf2f(p[(size_t)pix * 768]);
        s += v;
        if (v > top[9]) {
            int j = 9;
            while (j > 0 && top[j - 1] < v) { top[j] = top[j - 1]; --j; }
            top[j] = v;
        }
    }
    float tm = 0.f;
#pragma unroll
    for (int j = 0; j < 10; ++j) tm += top[j];
    pooled[i] = (s / 256.f + tm / 10.f) * 0.5f;
}

// ---------------------------------------------------------------------------
// Fused head: (16,768) -> 128 (bn1d+relu) -> 64 (bn1d+relu) -> 2.
// One workgroup, all intermediates in LDS.
// ---------------------------------------------------------------------------
__global__ void head_kernel(const float* __restrict__ pooled,
                            const float* __restrict__ g1w,
                            const float* __restrict__ bn1g,
                            const float* __restrict__ bn1b,
                            const float* __restrict__ g2w,
                            const float* __restrict__ bn2g,
                            const float* __restrict__ bn2b,
                            const float* __restrict__ g3w,
                            float* __restrict__ glout) {
    __shared__ float y1[16 * 128];
    __shared__ float y2[16 * 64];
    __shared__ float sc[128], sh[128];
    int t = threadIdx.x;
    for (int o = t; o < 16 * 128; o += 256) {
        int b = o / 128, j = o % 128;
        const float* pr = pooled + (size_t)b * 768;
        const float* wr = g1w + (size_t)j * 768;
        float s = 0.f;
        for (int d = 0; d < 768; ++d) s += pr[d] * wr[d];
        y1[o] = s;
    }
    __syncthreads();
    if (t < 128) {
        float s = 0.f, ss = 0.f;
        for (int b = 0; b < 16; ++b) { float v = y1[b * 128 + t]; s += v; ss += v * v; }
        float m = s / 16.f, var = ss / 16.f - m * m;
        float scl = bn1g[t] * rsqrtf(var + 1e-5f);
        sc[t] = scl; sh[t] = bn1b[t] - m * scl;
    }
    __syncthreads();
    for (int o = t; o < 16 * 128; o += 256) {
        int j = o % 128;
        float v = y1[o] * sc[j] + sh[j];
        y1[o] = v > 0.f ? v : 0.f;
    }
    __syncthreads();
    for (int o = t; o < 16 * 64; o += 256) {
        int b = o / 64, j = o % 64;
        const float* wr = g2w + (size_t)j * 128;
        float s = 0.f;
        for (int d = 0; d < 128; ++d) s += y1[b * 128 + d] * wr[d];
        y2[o] = s;
    }
    __syncthreads();
    if (t < 64) {
        float s = 0.f, ss = 0.f;
        for (int b = 0; b < 16; ++b) { float v = y2[b * 64 + t]; s += v; ss += v * v; }
        float m = s / 16.f, var = ss / 16.f - m * m;
        float scl = bn2g[t] * rsqrtf(var + 1e-5f);
        sc[t] = scl; sh[t] = bn2b[t] - m * scl;
    }
    __syncthreads();
    for (int o = t; o < 16 * 64; o += 256) {
        int j = o % 64;
        float v = y2[o] * sc[j] + sh[j];
        y2[o] = v > 0.f ? v : 0.f;
    }
    __syncthreads();
    if (t < 32) {
        int b = t / 2, o = t % 2;
        const float* wr = g3w + (size_t)o * 64;
        float s = 0.f;
        for (int d = 0; d < 64; ++d) s += y2[b * 64 + d] * wr[d];
        glout[b * 2 + o] = s;
    }
}

// ---------------------------------------------------------------------------
// Host orchestration
// ---------------------------------------------------------------------------
extern "C" void kernel_launch(void* const* d_in, const int* in_sizes, int n_in,
                              void* d_out, int out_size, void* d_ws, size_t ws_size,
                              hipStream_t stream) {
    (void)in_sizes; (void)n_in; (void)out_size; (void)ws_size;

    const float* query_patch  = (const float*)d_in[1];
    const float* prompt_patch = (const float*)d_in[3];
    const float* sbn_g  = (const float*)d_in[4];
    const float* sbn_b  = (const float*)d_in[5];
    const float* conv1w = (const float*)d_in[6];
    const float* bn1g   = (const float*)d_in[7];
    const float* bn1b   = (const float*)d_in[8];
    const float* ct1w   = (const float*)d_in[9];
    const float* ct1b   = (const float*)d_in[10];
    const float* conv2w = (const float*)d_in[11];
    const float* bn2g   = (const float*)d_in[12];
    const float* bn2b   = (const float*)d_in[13];
    const float* ct2w   = (const float*)d_in[14];
    const float* ct2b   = (const float*)d_in[15];
    const float* conv3w = (const float*)d_in[16];
    const float* conv3b = (const float*)d_in[17];
    const float* g1w    = (const float*)d_in[18];
    const float* gbn1g  = (const float*)d_in[19];
    const float* gbn1b  = (const float*)d_in[20];
    const float* g2w    = (const float*)d_in[21];
    const float* gbn2g  = (const float*)d_in[22];
    const float* gbn2b  = (const float*)d_in[23];
    const float* g3w    = (const float*)d_in[24];

    float* out = (float*)d_out;
    float* gl_out = out;                                     // (4,16,2)
    float* ls_out = out + 4 * 16 * 2;                        // (4,16,2,224,224)
    float* asc_out = ls_out + (size_t)4 * 16 * 2 * 224 * 224;// (4,16,1,224,224)

    char* ws = (char*)d_ws;
    size_t off = 0;
    auto alloc = [&](size_t bytes) -> void* {
        void* p = ws + off;
        off += (bytes + 255) & ~(size_t)255;
        return p;
    };
    unsigned short* W1t   = (unsigned short*)alloc((size_t)128 * 6912 * 2);
    unsigned short* W2t   = (unsigned short*)alloc((size_t)64 * 1152 * 2);
    unsigned short* Qbf   = (unsigned short*)alloc((size_t)4096 * 768 * 2);
    unsigned short* Pbf   = (unsigned short*)alloc((size_t)32768 * 768 * 2);
    float* alignArr       = (float*)alloc((size_t)4096 * 4);
    int*   idxArr         = (int*)alloc((size_t)4096 * 4);
    float* fus            = (float*)alloc((size_t)4096 * 768 * 4);
    unsigned short* fusBN = (unsigned short*)alloc((size_t)4096 * 768 * 2);
    float* scaleBuf       = (float*)alloc((size_t)768 * 4);
    float* shiftBuf       = (float*)alloc((size_t)768 * 4);
    float* c1             = (float*)alloc((size_t)4096 * 128 * 4);
    float* h1             = (float*)alloc((size_t)4096 * 128 * 4);
    unsigned short* d1bf  = (unsigned short*)alloc((size_t)16384 * 128 * 2);
    float* c2             = (float*)alloc((size_t)16384 * 64 * 4);
    float* h2             = (float*)alloc((size_t)16384 * 64 * 4);
    unsigned short* d2bf  = (unsigned short*)alloc((size_t)65536 * 64 * 2);
    float* sm             = (float*)alloc((size_t)65536 * 2 * 4);
    float* pooled         = (float*)alloc((size_t)16 * 768 * 4);

    auto nb = [](size_t total) -> unsigned { return (unsigned)((total + 255) / 256); };

    // Weight transforms (shared across layers).
    wtrans_kernel<<<nb((size_t)128 * 768 * 9), 256, 0, stream>>>(conv1w, W1t, 128, 768);
    wtrans_kernel<<<nb((size_t)64 * 128 * 9), 256, 0, stream>>>(conv2w, W2t, 64, 128);

    for (int l = 0; l < 4; ++l) {
        const float* qp = query_patch  + (size_t)l * 16 * 257 * 768;
        const float* pp = prompt_patch + (size_t)l * 16 * 8 * 257 * 768;

        qnorm_kernel<<<4096, 256, 0, stream>>>(qp, Qbf);
        pnorm_kernel<<<32768, 256, 0, stream>>>(pp, Pbf);

        cost_wmma_kernel<<<dim3(16, 16), 256, 0, stream>>>(Qbf, Pbf, alignArr, idxArr);

        fusion_kernel<<<nb((size_t)16 * 256 * 768), 256, 0, stream>>>(Qbf, Pbf, idxArr, fus);

        // shared BN (per-layer gamma/beta), writes bf16 for conv1 + pooling
        bnstats_kernel<<<768, 256, 0, stream>>>(fus, 768, 4096,
                                                sbn_g + l * 768, sbn_b + l * 768,
                                                scaleBuf, shiftBuf);
        bnapply_kernel<<<nb((size_t)16 * 256 * 768), 256, 0, stream>>>(
            fus, scaleBuf, shiftBuf, 768, (size_t)16 * 256 * 768, fusBN, nullptr, 0);

        // conv1 (768 -> 128, 16x16) — WMMA implicit GEMM
        conv3x3_wmma_kernel<<<dim3(256, 8), 32, 0, stream>>>(fusBN, W1t, c1, 16, 768, 128);
        bnstats_kernel<<<128, 256, 0, stream>>>(c1, 128, 4096, bn1g, bn1b, scaleBuf, shiftBuf);
        bnapply_kernel<<<nb((size_t)4096 * 128), 256, 0, stream>>>(
            c1, scaleBuf, shiftBuf, 128, (size_t)4096 * 128, nullptr, h1, 1);

        // deconv1 -> (16,32,32,128) bf16
        deconv_kernel<<<nb((size_t)16 * 32 * 32 * 128), 256, 0, stream>>>(
            h1, ct1w, ct1b, d1bf, 16, 128, 128);

        // conv2 (128 -> 64, 32x32) — WMMA implicit GEMM
        conv3x3_wmma_kernel<<<dim3(1024, 4), 32, 0, stream>>>(d1bf, W2t, c2, 32, 128, 64);
        bnstats_kernel<<<64, 256, 0, stream>>>(c2, 64, 16384, bn2g, bn2b, scaleBuf, shiftBuf);
        bnapply_kernel<<<nb((size_t)16384 * 64), 256, 0, stream>>>(
            c2, scaleBuf, shiftBuf, 64, (size_t)16384 * 64, nullptr, h2, 1);

        // deconv2 -> (16,64,64,64) bf16
        deconv_kernel<<<nb((size_t)16 * 64 * 64 * 64), 256, 0, stream>>>(
            h2, ct2w, ct2b, d2bf, 32, 64, 64);

        // conv3 1x1 + softmax
        conv3sm_kernel<<<nb((size_t)16 * 4096), 256, 0, stream>>>(d2bf, conv3w, conv3b, sm);

        // ls: resize (16,2,64,64) -> (16,2,224,224)
        resize_kernel<<<nb((size_t)16 * 2 * 224 * 224), 256, 0, stream>>>(
            sm, ls_out + (size_t)l * 16 * 2 * 224 * 224,
            16, 2, 64, (long)4096 * 2, 1L, 2L);

        // asc: resize (16,1,16,16) -> (16,1,224,224)
        resize_kernel<<<nb((size_t)16 * 224 * 224), 256, 0, stream>>>(
            alignArr, asc_out + (size_t)l * 16 * 224 * 224,
            16, 1, 16, 256L, 0L, 1L);

        // global head
        pooled_kernel<<<48, 256, 0, stream>>>(fusBN, pooled);
        head_kernel<<<1, 256, 0, stream>>>(pooled, g1w, gbn1g, gbn1b,
                                           g2w, gbn2g, gbn2b, g3w,
                                           gl_out + (size_t)l * 32);
    }
}
